// LRUEmbedding_47588237640302
// MI455X (gfx1250) — compile-verified, matrix operand order checked by reading, not run on hardware
//
#include <hip/hip_runtime.h>

typedef __attribute__((ext_vector_type(16))) _Float16 v16h;
typedef __attribute__((ext_vector_type(8)))  _Float16 v8h;
typedef __attribute__((ext_vector_type(8)))  float    v8f;

union HFrag { v16h f; v8h h[2]; };

#define BATCH  8
#define SEQ    4096
#define DIN    256
#define S1     256
#define S2     512
#define NJ     1024        // 2*S2 real columns: [re | im]
#define MROWS  (BATCH*SEQ) // 32768
#define CHUNKS 64
#define CLEN   64          // CHUNKS*CLEN == SEQ

// ---- workspace byte offsets (all 16B aligned) ----
#define WS_W1T   0u          // [NJ][DIN]  f16  (B_norm, transposed, 512 KB)
#define WS_W2T   524288u     // [DIN][NJ]  f16  (C packed, transposed, 512 KB)
#define WS_LAMR  1048576u    // [S2] f32
#define WS_LAMI  1050624u
#define WS_APR   1052672u    // lam^CLEN
#define WS_API   1054720u
#define WS_SUMR  1056768u    // [B][CHUNKS][S2] f32 (1 MB)
#define WS_SUMI  2105344u
#define WS_UBN   3153920u    // [MROWS][NJ] f16 (67 MB) — overwritten in place by x
#define WS_U16   70262784u   // [MROWS][DIN] f16 (16 MB)

// CDNA5 async copy: LDS[ldsOff] = MEM[sbase + voff], 16 bytes per lane.
// Tracked by ASYNCcnt (loads complete in order).
__device__ __forceinline__ void async_b128(unsigned ldsOff, unsigned voff,
                                           unsigned long long sbase) {
  asm volatile("global_load_async_to_lds_b128 %0, %1, %2"
               :: "v"(ldsOff), "v"(voff), "s"(sbase) : "memory");
}
__device__ __forceinline__ void wait_async_le4() {
  asm volatile("s_wait_asynccnt 0x4" ::: "memory");
}
__device__ __forceinline__ void wait_async_0() {
  asm volatile("s_wait_asynccnt 0x0" ::: "memory");
}

// ---------------------------------------------------------------------------
// Phase 0a: weight prep + eigenvalue prep
// ---------------------------------------------------------------------------
__global__ __launch_bounds__(256) void lru_prep(
    const float* __restrict__ Bre, const float* __restrict__ Bim,
    const float* __restrict__ Cre, const float* __restrict__ Cim,
    const float* __restrict__ log_nu, const float* __restrict__ log_th,
    const float* __restrict__ log_g,
    _Float16* __restrict__ W1t, _Float16* __restrict__ W2t,
    float* __restrict__ lamR, float* __restrict__ lamI,
    float* __restrict__ ApR, float* __restrict__ ApI) {
  int idx = blockIdx.x * 256 + threadIdx.x;   // 0 .. 262143
  {
    int j = idx >> 8, k = idx & 255;
    float w;
    if (j < S2) { w = Bre[j * DIN + k] * expf(log_g[j]); }
    else        { int jj = j - S2; w = Bim[jj * DIN + k] * expf(log_g[jj]); }
    W1t[(size_t)j * DIN + k] = (_Float16)w;
  }
  {
    int d = idx >> 10, p = idx & 1023;
    float w = (p < S2) ? Cre[d * S2 + p] : -Cim[d * S2 + (p - S2)];
    W2t[(size_t)d * NJ + p] = (_Float16)w;
  }
  if (idx < S2) {
    float nu = expf(log_nu[idx]);
    float th = expf(log_th[idx]);
    float r  = expf(-nu);
    float lr = r * cosf(th), li = r * sinf(th);
    lamR[idx] = lr; lamI[idx] = li;
    float ar = lr, ai = li;
#pragma unroll
    for (int q = 0; q < 6; ++q) {           // lam^(2^6) = lam^64
      float nr = ar * ar - ai * ai;
      float ni = 2.f * ar * ai;
      ar = nr; ai = ni;
    }
    ApR[idx] = ar; ApI[idx] = ai;
  }
}

// ---------------------------------------------------------------------------
// Phase 0b: input f32 -> f16 (one pass; halves GEMM1 A-traffic and makes the
// GEMM tiles pure f16 so the async-to-LDS path can stage them directly).
// ---------------------------------------------------------------------------
__global__ __launch_bounds__(256) void lru_cvt(const float* __restrict__ U,
                                               _Float16* __restrict__ U16) {
  size_t idx = (size_t)(blockIdx.x * 256 + threadIdx.x) * 8;
  float4 a0 = *(const float4*)(U + idx);
  float4 a1 = *(const float4*)(U + idx + 4);
  v8h hv;
  hv[0] = (_Float16)a0.x; hv[1] = (_Float16)a0.y;
  hv[2] = (_Float16)a0.z; hv[3] = (_Float16)a0.w;
  hv[4] = (_Float16)a1.x; hv[5] = (_Float16)a1.y;
  hv[6] = (_Float16)a1.z; hv[7] = (_Float16)a1.w;
  *(v8h*)(U16 + idx) = hv;
}

// ---------------------------------------------------------------------------
// Shared WMMA GEMM body: C[128x128] tile, 8 waves (2x4), wave tile 64x32.
// A: [M][KD] f16 row-major; B(T): [N][KD] f16 row-major (K contiguous).
// Double-buffered LDS, async global->LDS staging.
// acc indexed [mf 0..3][nf 0..1].
// ---------------------------------------------------------------------------
template <int KD>
__device__ __forceinline__ void gemm_tile(
    const _Float16* __restrict__ A, const _Float16* __restrict__ Bt,
    int rowBase, int colBase, int tid, v8f acc[4][2]) {
  __shared__ __align__(16) _Float16 lds[4 * 4096];  // A:buf0,buf1 | B:buf0,buf1
  const unsigned ldsBase = (unsigned)(size_t)(&lds[0]);

  const int wid = tid >> 5, lane = tid & 31;
  const int wm = wid >> 2, wn = wid & 3;      // 2 x 4 wave grid
  const int l16 = lane & 15, half = lane >> 4;

  // per-thread staging chunks: lin = tid + i*256 -> (r, g); 16B per chunk
  int r0 = tid >> 2,            g0 = tid & 3;
  int r1 = (tid + 256) >> 2,    g1 = (tid + 256) & 3;
  const unsigned la0 = (unsigned)(r0 * 64 + g0 * 16);          // LDS byte off in A tile
  const unsigned la1 = (unsigned)(r1 * 64 + g1 * 16);
  const unsigned aoff0 = (unsigned)(((rowBase + r0) * KD + g0 * 8) * 2);
  const unsigned aoff1 = (unsigned)(((rowBase + r1) * KD + g1 * 8) * 2);
  const unsigned boff0 = (unsigned)(((colBase + r0) * KD + g0 * 8) * 2);
  const unsigned boff1 = (unsigned)(((colBase + r1) * KD + g1 * 8) * 2);
  const unsigned long long abase = (unsigned long long)A;
  const unsigned long long bbase = (unsigned long long)Bt;

  const int NSTEP = KD / 32;
  // preload buffer 0 (kt = 0)
  async_b128(ldsBase + la0,         aoff0, abase);
  async_b128(ldsBase + la1,         aoff1, abase);
  async_b128(ldsBase + 16384 + la0, boff0, bbase);
  async_b128(ldsBase + 16384 + la1, boff1, bbase);

  for (int step = 0; step < NSTEP; ++step) {
    const unsigned cur = (unsigned)(step & 1);
    __syncthreads();                       // next buffer free to overwrite
    if (step + 1 < NSTEP) {
      const unsigned nxt = cur ^ 1u;
      const unsigned kb = (unsigned)((step + 1) * 64);   // kt*2 bytes
      async_b128(ldsBase + nxt * 8192 + la0,         aoff0 + kb, abase);
      async_b128(ldsBase + nxt * 8192 + la1,         aoff1 + kb, abase);
      async_b128(ldsBase + 16384 + nxt * 8192 + la0, boff0 + kb, bbase);
      async_b128(ldsBase + 16384 + nxt * 8192 + la1, boff1 + kb, bbase);
      if (step + 2 < NSTEP) {              // warm L2/WGP$ for the tile after
        __builtin_prefetch((const char*)A  + aoff0 + kb + 64, 0, 3);
        __builtin_prefetch((const char*)Bt + boff0 + kb + 64, 0, 3);
      }
      wait_async_le4();                    // group `step` done (in-order)
    } else {
      wait_async_0();
    }
    __syncthreads();                       // all waves' tiles visible

    const _Float16* bufA = &lds[cur * 4096];
    const _Float16* bufB = &lds[8192 + cur * 4096];
    HFrag fa[4], fb[2];
#pragma unroll
    for (int mf = 0; mf < 4; ++mf) {
      int rr = wm * 64 + mf * 16 + l16;
      fa[mf].h[0] = *(const v8h*)&bufA[rr * 32 + half * 8];
      fa[mf].h[1] = *(const v8h*)&bufA[rr * 32 + 16 + half * 8];
    }
#pragma unroll
    for (int nf = 0; nf < 2; ++nf) {
      int nn = wn * 32 + nf * 16 + l16;
      fb[nf].h[0] = *(const v8h*)&bufB[nn * 32 + half * 8];
      fb[nf].h[1] = *(const v8h*)&bufB[nn * 32 + 16 + half * 8];
    }
#pragma unroll
    for (int mf = 0; mf < 4; ++mf)
#pragma unroll
      for (int nf = 0; nf < 2; ++nf)
        acc[mf][nf] = __builtin_amdgcn_wmma_f32_16x16x32_f16(
            false, fa[mf].f, false, fb[nf].f, (short)0, acc[mf][nf], false, false);
  }
}

// ---------------------------------------------------------------------------
// GEMM1: ubn[m][j] = sum_k u16[m][k] * W1t[j][k]  (M=32768, K=256, N=1024)
// ---------------------------------------------------------------------------
__global__ __launch_bounds__(256) void lru_gemm1(
    const _Float16* __restrict__ U16, const _Float16* __restrict__ W1t,
    _Float16* __restrict__ ubn) {
  const int tid = threadIdx.x;
  const int gm = blockIdx.x & 255;        // M/128
  const int gn = blockIdx.x >> 8;         // NJ/128 = 8
  const int rowBase = gm << 7, colBase = gn << 7;
  v8f acc[4][2];
  v8f zero = {};
#pragma unroll
  for (int mf = 0; mf < 4; ++mf)
#pragma unroll
    for (int nf = 0; nf < 2; ++nf) acc[mf][nf] = zero;

  gemm_tile<DIN>(U16, W1t, rowBase, colBase, tid, acc);

  const int wid = tid >> 5, lane = tid & 31;
  const int wm = wid >> 2, wn = wid & 3;
  const int l16 = lane & 15, half = lane >> 4;
#pragma unroll
  for (int mf = 0; mf < 4; ++mf)
#pragma unroll
    for (int nf = 0; nf < 2; ++nf)
#pragma unroll
      for (int v = 0; v < 8; ++v) {
        int m = rowBase + wm * 64 + mf * 16 + v + half * 8;
        int n = colBase + wn * 32 + nf * 16 + l16;
        ubn[(size_t)m * NJ + n] = (_Float16)acc[mf][nf][v];
      }
}

// ---------------------------------------------------------------------------
// Scan pass 1: per (b, chunk, s) reduce chunk with zero init.
// ---------------------------------------------------------------------------
__global__ __launch_bounds__(256) void lru_scan_reduce(
    const _Float16* __restrict__ ubn, const float* __restrict__ lamR,
    const float* __restrict__ lamI, float* __restrict__ sumR,
    float* __restrict__ sumI) {
  int idx = blockIdx.x * 256 + threadIdx.x;   // b*32768 + c*512 + s
  int s = idx & 511, c = (idx >> 9) & 63, b = idx >> 15;
  float lr = lamR[s], li = lamI[s];
  float xr = 0.f, xi = 0.f;
  const bool fwd = (s < S1);
  const int tau0 = c * CLEN;
  for (int q = 0; q < CLEN; ++q) {
    int tau = tau0 + q;
    int t = fwd ? tau : (SEQ - 1 - tau);
    size_t base = ((size_t)(b * SEQ + t)) * NJ + s;
    float ur = (float)ubn[base];
    float ui = (float)ubn[base + S2];
    float nr = lr * xr - li * xi + ur;
    float ni = lr * xi + li * xr + ui;
    xr = nr; xi = ni;
  }
  sumR[idx] = xr; sumI[idx] = xi;
}

// ---------------------------------------------------------------------------
// Scan pass 2: per (b, s) serial prefix over chunk summaries; carry0 = state.
// ---------------------------------------------------------------------------
__global__ __launch_bounds__(256) void lru_scan_prefix(
    const float* __restrict__ stR, const float* __restrict__ stI,
    const float* __restrict__ ApR, const float* __restrict__ ApI,
    float* __restrict__ sumR, float* __restrict__ sumI) {
  int idx = blockIdx.x * 256 + threadIdx.x;   // b*512 + s
  int s = idx & 511, b = idx >> 9;
  float ar = ApR[s], ai = ApI[s];
  float cr = stR[b * S2 + s], ci = stI[b * S2 + s];
  for (int c = 0; c < CHUNKS; ++c) {
    int j = b * (CHUNKS * S2) + c * S2 + s;
    float pr = sumR[j], pi = sumI[j];
    sumR[j] = cr; sumI[j] = ci;               // carry-in for chunk c
    float nr = ar * cr - ai * ci + pr;        // x_end = lam^L * carry + p
    float ni = ar * ci + ai * cr + pi;
    cr = nr; ci = ni;
  }
}

// ---------------------------------------------------------------------------
// Scan pass 3: seeded chunk apply; overwrite ubn in place with x (f16).
// ---------------------------------------------------------------------------
__global__ __launch_bounds__(256) void lru_scan_apply(
    _Float16* __restrict__ ubn, const float* __restrict__ lamR,
    const float* __restrict__ lamI, const float* __restrict__ sumR,
    const float* __restrict__ sumI) {
  int idx = blockIdx.x * 256 + threadIdx.x;
  int s = idx & 511, c = (idx >> 9) & 63, b = idx >> 15;
  float lr = lamR[s], li = lamI[s];
  float xr = sumR[idx], xi = sumI[idx];
  const bool fwd = (s < S1);
  const int tau0 = c * CLEN;
  for (int q = 0; q < CLEN; ++q) {
    int tau = tau0 + q;
    int t = fwd ? tau : (SEQ - 1 - tau);
    size_t base = ((size_t)(b * SEQ + t)) * NJ + s;
    float ur = (float)ubn[base];
    float ui = (float)ubn[base + S2];
    float nr = lr * xr - li * xi + ur;
    float ni = lr * xi + li * xr + ui;
    xr = nr; xi = ni;
    ubn[base]      = (_Float16)xr;            // x_re -> col s
    ubn[base + S2] = (_Float16)xi;            // x_im -> col 512+s
  }
}

// ---------------------------------------------------------------------------
// GEMM2: y[m][d] = sum_p x[m][p]*W2t[d][p] + u[m][d]*D[d]  (K=1024, N=256)
// ---------------------------------------------------------------------------
__global__ __launch_bounds__(256) void lru_gemm2(
    const _Float16* __restrict__ X, const _Float16* __restrict__ W2t,
    const float* __restrict__ U, const float* __restrict__ Dv,
    float* __restrict__ Y) {
  const int tid = threadIdx.x;
  const int gm = blockIdx.x & 255;        // M/128
  const int gn = blockIdx.x >> 8;         // DIN/128 = 2
  const int rowBase = gm << 7, colBase = gn << 7;
  v8f acc[4][2];
  v8f zero = {};
#pragma unroll
  for (int mf = 0; mf < 4; ++mf)
#pragma unroll
    for (int nf = 0; nf < 2; ++nf) acc[mf][nf] = zero;

  gemm_tile<NJ>(X, W2t, rowBase, colBase, tid, acc);

  const int wid = tid >> 5, lane = tid & 31;
  const int wm = wid >> 2, wn = wid & 3;
  const int l16 = lane & 15, half = lane >> 4;
#pragma unroll
  for (int mf = 0; mf < 4; ++mf)
#pragma unroll
    for (int nf = 0; nf < 2; ++nf) {
      int d = colBase + wn * 32 + nf * 16 + l16;
      float dv = Dv[d];
#pragma unroll
      for (int v = 0; v < 8; ++v) {
        int m = rowBase + wm * 64 + mf * 16 + v + half * 8;
        float val = acc[mf][nf][v] + U[(size_t)m * DIN + d] * dv;
        Y[(size_t)m * DIN + d] = val;
      }
    }
}

// ---------------------------------------------------------------------------
extern "C" void kernel_launch(void* const* d_in, const int* in_sizes, int n_in,
                              void* d_out, int out_size, void* d_ws, size_t ws_size,
                              hipStream_t stream) {
  (void)in_sizes; (void)n_in; (void)out_size; (void)ws_size;
  const float* input  = (const float*)d_in[0];
  const float* st_re  = (const float*)d_in[1];
  const float* st_im  = (const float*)d_in[2];
  const float* log_nu = (const float*)d_in[3];
  const float* log_th = (const float*)d_in[4];
  const float* B_re   = (const float*)d_in[5];
  const float* B_im   = (const float*)d_in[6];
  const float* C_re   = (const float*)d_in[7];
  const float* C_im   = (const float*)d_in[8];
  const float* Dv     = (const float*)d_in[9];
  const float* log_g  = (const float*)d_in[10];
  float* Y = (float*)d_out;

  char* ws = (char*)d_ws;
  _Float16* W1t  = (_Float16*)(ws + WS_W1T);
  _Float16* W2t  = (_Float16*)(ws + WS_W2T);
  float*    lamR = (float*)(ws + WS_LAMR);
  float*    lamI = (float*)(ws + WS_LAMI);
  float*    ApR  = (float*)(ws + WS_APR);
  float*    ApI  = (float*)(ws + WS_API);
  float*    sumR = (float*)(ws + WS_SUMR);
  float*    sumI = (float*)(ws + WS_SUMI);
  _Float16* ubn  = (_Float16*)(ws + WS_UBN);
  _Float16* U16  = (_Float16*)(ws + WS_U16);

  lru_prep<<<1024, 256, 0, stream>>>(B_re, B_im, C_re, C_im, log_nu, log_th,
                                     log_g, W1t, W2t, lamR, lamI, ApR, ApI);
  lru_cvt<<<4096, 256, 0, stream>>>(input, U16);
  lru_gemm1<<<2048, 256, 0, stream>>>(U16, W1t, ubn);           // 256 x 8 tiles
  lru_scan_reduce<<<1024, 256, 0, stream>>>(ubn, lamR, lamI, sumR, sumI);
  lru_scan_prefix<<<16, 256, 0, stream>>>(st_re, st_im, ApR, ApI, sumR, sumI);
  lru_scan_apply<<<1024, 256, 0, stream>>>(ubn, lamR, lamI, sumR, sumI);
  lru_gemm2<<<512, 256, 0, stream>>>(ubn, W2t, input, Dv, Y);   // 256 x 2 tiles
}
